// Solver_old_69106023792719
// MI455X (gfx1250) — matrix operand order repaired and checked by reading, not compiled, not run on hardware
//
#include <hip/hip_runtime.h>
#include <hip/hip_bf16.h>
#include <math.h>

typedef __attribute__((ext_vector_type(16))) _Float16 v16h;
typedef __attribute__((ext_vector_type(8)))  _Float16 v8h;
typedef __attribute__((ext_vector_type(8)))  float    v8f;

#define N_DIM 466
#define P_DIM 4
#define Z_DIM 64
#define T_PTS 1024
#define HID   173
#define NP    470

#define PACK  2080              // 64*65/2 packed symmetric (j<=k) entries
#define KTOT  2144              // PACK + 64 (A folded in)
#define KPAD  2176              // padded to 68*32 so all 16 waves do 17 const-trip WMMAs
#define KQ    17                // K-tiles per wave (68 tiles / 4 K-quarters)

// d_out layout (floats): n_s [1024*466] | z_s [1024*64] | status [1]
#define NS_SZ   (T_PTS * N_DIM)     // 477184
#define ZS_OFF  NS_SZ
#define ZS_SZ   (T_PTS * Z_DIM)     // 65536
#define ST_OFF  (ZS_OFF + ZS_SZ)    // 542720

// workspace layout (bytes)
#define WS_BS16  0                  // 64*2176 f16          = 278528
#define WS_WD1T  278528             // 192*64 f16           = 24576
#define WS_WD2T  303104             // 480*192 f16          = 184320
#define WS_ZS16  487424             // 1024*64 f16          = 131072
#define WS_HD16  618496             // 1024*192 f16         = 393216
#define WS_Z0    1011712            // 64 f32

// ODE kernel dynamic LDS layout (bytes)
#define L_BS   0                    // 64*2176 halfs = 278528
#define L_Q    278528               // 2176 halfs    = 4352
#define L_JK   282880               // 2080 uint     = 8320
#define L_F32  291200               // f32 scratch
#define SMEM_BYTES 294912

__device__ __forceinline__ v8f wmma_f16(v16h a, v16h b, v8f c) {
  return __builtin_amdgcn_wmma_f32_16x16x32_f16(false, a, false, b, (short)0, c, false, false);
}

// ---------------- prep kernels ----------------

// Bs[i][t]: t<PACK: symmetrized B (j<k: B[i,j,k]+B[i,k,j]; j==k: B[i,j,j]);
// PACK<=t<KTOT: A[i][t-PACK]; t>=KTOT: 0 (K padding).
// Then g(z)_i = sum_t Bs[i,t]*q[t] with q = (z_j z_k | z | 0).
__global__ void prep_bsym_kernel(const float* __restrict__ B, const float* __restrict__ A,
                                 _Float16* __restrict__ Bs) {
  int e = blockIdx.x * 256 + threadIdx.x;
  if (e >= 64 * KPAD) return;
  int i = e / KPAD, t = e - i * KPAD;
  float v;
  if (t < PACK) {
    int j = 0, rem = t, len = 64;
    while (rem >= len) { rem -= len; ++j; --len; }
    int k = j + rem;
    v = (j == k) ? B[(i * 64 + j) * 64 + j]
                 : (B[(i * 64 + j) * 64 + k] + B[(i * 64 + k) * 64 + j]);
  } else if (t < KTOT) {
    v = A[i * 64 + (t - PACK)];
  } else {
    v = 0.0f;
  }
  Bs[e] = (_Float16)v;
}

// W_d1 [64,HID] -> f16 transposed/padded [192][64] (n-major rows)
__global__ void prep_wd1_kernel(const float* __restrict__ W, _Float16* __restrict__ WT) {
  int e = blockIdx.x * 256 + threadIdx.x;
  if (e >= 192 * 64) return;
  int n = e / 64, k = e - n * 64;
  WT[e] = (_Float16)((n < HID) ? W[k * HID + n] : 0.0f);
}

// W_d2 [HID,466] -> f16 transposed/padded [480][192]
__global__ void prep_wd2_kernel(const float* __restrict__ W, _Float16* __restrict__ WT) {
  int e = blockIdx.x * 256 + threadIdx.x;
  if (e >= 480 * 192) return;
  int n = e / 192, k = e - n * 192;
  WT[e] = (_Float16)((n < N_DIM && k < HID) ? W[k * N_DIM + n] : 0.0f);
}

// ---------------- encoder ----------------

__global__ __launch_bounds__(256) void encoder_kernel(
    const float* __restrict__ n0, const float* __restrict__ p,
    const float* __restrict__ We1, const float* __restrict__ be1,
    const float* __restrict__ We2, const float* __restrict__ be2,
    float* __restrict__ z0out) {
  __shared__ float x0[NP];
  __shared__ float hh[HID];
  int tid = threadIdx.x;
  for (int j = tid; j < NP; j += 256) x0[j] = (j < P_DIM) ? p[j] : n0[j - P_DIM];
  __syncthreads();
  if (tid < HID) {
    float acc = be1[tid];
    for (int j = 0; j < NP; ++j) acc = fmaf(x0[j], We1[j * HID + tid], acc);
    hh[tid] = tanhf(acc);
  }
  __syncthreads();
  if (tid < Z_DIM) {
    float acc = be2[tid];
    for (int j = 0; j < HID; ++j) acc = fmaf(hh[j], We2[j * Z_DIM + tid], acc);
    z0out[tid] = tanhf(acc);
  }
}

// ---------------- ODE solver (persistent 512-thread workgroup, WMMA RHS) ----------------

// Computes partial results of g(zc) into part[4][64]; caller combines.
__device__ __forceinline__ void g_eval_parts(_Float16* Bs, _Float16* q, const unsigned* jk,
                                             const float* zc, float* part, int tid) {
  // build q = [z_j*z_k (j<=k) | z | 0-pad]
  for (int t = tid; t < KPAD; t += 512) {
    float v;
    if (t < PACK)      { unsigned u = jk[t]; v = zc[u & 0xffffu] * zc[u >> 16]; }
    else if (t < KTOT) { v = zc[t - PACK]; }
    else               { v = 0.0f; }
    q[t] = (_Float16)v;
  }
  __syncthreads();
  int lane = tid & 31, wave = tid >> 5;
  int mtile = wave & 3, kq = wave >> 2;          // 16 waves: 4 M-tiles x 4 K-quarters
  int m    = mtile * 16 + (lane & 15);
  int ksub = (lane >> 4) * 8;                    // A-operand sub-offset in 2-half units (k = 16h + s)
  const _Float16* brow = Bs + m * KPAD + kq * (KQ * 32) + ksub * 2;
  const _Float16* qrow = q + kq * (KQ * 32) + ksub * 2;
  v8f acc = {};
#pragma unroll
  for (int u = 0; u < KQ; ++u) {
    v8h a0 = *(const v8h*)(brow + u * 32);
    v8h a1 = *(const v8h*)(brow + u * 32 + 8);
    v16h av = __builtin_shufflevector(a0, a1, 0,1,2,3,4,5,6,7,8,9,10,11,12,13,14,15);
    // broadcast-B: every N column = q slice -> every D column = result
    v16h bv = *(const v16h*)(qrow + u * 32);
    acc = wmma_f16(av, bv, acc);
  }
  if (lane == 0 || lane == 16) {
    int rb = mtile * 16 + ((lane >> 4) ? 8 : 0);
#pragma unroll
    for (int r = 0; r < 8; ++r) part[kq * 64 + rb + r] = acc[r];
  }
  __syncthreads();
}

__device__ __forceinline__ float part_sum(const float* part, int i) {
  return (part[i] + part[64 + i]) + (part[128 + i] + part[192 + i]);
}

__global__ __launch_bounds__(512) void ode_kernel(const float* __restrict__ tstep,
                                                  const float* __restrict__ z0,
                                                  const _Float16* __restrict__ Bs_g,
                                                  float* __restrict__ zs_out) {
  extern __shared__ char smem[];
  _Float16* Bs = (_Float16*)(smem + L_BS);
  _Float16* q  = (_Float16*)(smem + L_Q);
  unsigned* jk = (unsigned*)(smem + L_JK);
  float* fb    = (float*)(smem + L_F32);
  float* y   = fb;          // 64
  float* y5  = fb + 64;     // 64
  float* zc  = fb + 128;    // 64
  float* kk  = fb + 192;    // 7*64
  float* part= fb + 640;    // 4*64
  float* red = fb + 896;    // 16
  float* scal= fb + 912;    // [0]=h [1]=t [2]=dt [3]=ok [4]=done
  int tid = threadIdx.x;

  { // load symmetrized B (+A, +pad) into LDS once: ~278 KB, resident for whole solve
    const uint4* src = (const uint4*)Bs_g;
    uint4* dst = (uint4*)Bs;
    for (int e = tid; e < (64 * KPAD) / 8; e += 512) dst[e] = src[e];
  }
  for (int t = tid; t < PACK; t += 512) {  // (j,k) lookup for packed index
    int j = 0, rem = t, len = 64;
    while (rem >= len) { rem -= len; ++j; --len; }
    jk[t] = (unsigned)j | ((unsigned)(j + rem) << 16);
  }
  if (tid < Z_DIM) { y[tid] = z0[tid]; zs_out[tid] = z0[tid]; }
  if (tid == 0) scal[2] = fmaxf(tstep[1] - tstep[0], 1e-6f);
  __syncthreads();

  float* k1 = kk;       float* k2 = kk + 64;  float* k3 = kk + 128;
  float* k4 = kk + 192; float* k5 = kk + 256; float* k6 = kk + 320; float* k7 = kk + 384;

  for (int it = 0; it < T_PTS - 1; ++it) {
    float t1 = tstep[it + 1];
    if (tid == 0) scal[1] = tstep[it];
    __syncthreads();
    for (int st = 0; st < 6; ++st) {
      if (tid == 0) {
        float tc = scal[1], dt = scal[2];
        scal[4] = (tc >= t1 - 1e-12f) ? 1.f : 0.f;
        scal[0] = fmaxf(fminf(dt, t1 - tc), 0.f);
      }
      __syncthreads();
      g_eval_parts(Bs, q, jk, y, part, tid);           // k1 = g(y)
      float h = scal[0];
      if (tid < Z_DIM) {
        float v = part_sum(part, tid); k1[tid] = v;
        zc[tid] = y[tid] + h * (0.2f * v);
      }
      __syncthreads();
      g_eval_parts(Bs, q, jk, zc, part, tid);          // k2
      if (tid < Z_DIM) {
        float v = part_sum(part, tid); k2[tid] = v;
        zc[tid] = y[tid] + h * (0.075f * k1[tid] + 0.225f * v);
      }
      __syncthreads();
      g_eval_parts(Bs, q, jk, zc, part, tid);          // k3
      if (tid < Z_DIM) {
        float v = part_sum(part, tid); k3[tid] = v;
        zc[tid] = y[tid] + h * (0.9777777778f * k1[tid] - 3.7333333333f * k2[tid]
                              + 3.5555555556f * v);
      }
      __syncthreads();
      g_eval_parts(Bs, q, jk, zc, part, tid);          // k4
      if (tid < Z_DIM) {
        float v = part_sum(part, tid); k4[tid] = v;
        zc[tid] = y[tid] + h * (2.9525986893f * k1[tid] - 11.5957933357f * k2[tid]
                              + 9.8228928855f * k3[tid] - 0.2908093279f * v);
      }
      __syncthreads();
      g_eval_parts(Bs, q, jk, zc, part, tid);          // k5
      if (tid < Z_DIM) {
        float v = part_sum(part, tid); k5[tid] = v;
        zc[tid] = y[tid] + h * (2.8462752525f * k1[tid] - 10.7575757576f * k2[tid]
                              + 8.9064227150f * k3[tid] + 0.2784090909f * k4[tid]
                              - 0.2735313036f * v);
      }
      __syncthreads();
      g_eval_parts(Bs, q, jk, zc, part, tid);          // k6
      if (tid < Z_DIM) {
        float v = part_sum(part, tid); k6[tid] = v;
        float y5v = y[tid] + h * (0.0911458333f * k1[tid] + 0.4492362983f * k3[tid]
                                + 0.6510416667f * k4[tid] - 0.3223761792f * k5[tid]
                                + 0.1309523810f * v);
        y5[tid] = y5v; zc[tid] = y5v;
      }
      __syncthreads();
      g_eval_parts(Bs, q, jk, zc, part, tid);          // k7 = g(y5)
      if (tid < Z_DIM) {
        float k7v = part_sum(part, tid);
        float err = h * (0.0012326389f * k1[tid] - 0.0042527703f * k3[tid]
                       + 0.0369791667f * k4[tid] - 0.0508637836f * k5[tid]
                       + 0.0419047619f * k6[tid] - 0.0250000000f * k7v);
        (void)k7;
        float s = 1e-5f + 1e-2f * fmaxf(fabsf(y[tid]), fabsf(y5[tid]));
        float r = err / s;
        float v = r * r;
        // wave32 tree reduction (threads 0..63 = waves 0,1)
#pragma unroll
        for (int off = 16; off > 0; off >>= 1) v += __shfl_xor(v, off, 32);
        if ((tid & 31) == 0) red[tid >> 5] = v;
      }
      __syncthreads();
      if (tid == 0) {
        float errn = sqrtf((red[0] + red[1]) * (1.0f / 64.0f));
        float done = scal[4];
        int ok = (errn <= 1.0f) && (done == 0.f);
        scal[3] = ok ? 1.f : 0.f;
        if (ok) scal[1] += h;
        float fac = 0.9f * powf(errn + 1e-10f, -0.2f);
        fac = fminf(fmaxf(fac, 0.2f), 10.f);
        if (done == 0.f) scal[2] = fmaxf(h * fac, 1e-9f);
      }
      __syncthreads();
      if (tid < Z_DIM && scal[3] != 0.f) y[tid] = y5[tid];
      __syncthreads();
    }
    if (tid < Z_DIM) zs_out[(it + 1) * Z_DIM + tid] = y[tid];
    __syncthreads();
  }
}

// ---------------- decoder ----------------

__global__ void conv_zs_kernel(const float* __restrict__ zs, _Float16* __restrict__ zs16,
                               int* __restrict__ status) {
  int e = blockIdx.x * 256 + threadIdx.x;
  if (e < T_PTS * Z_DIM) zs16[e] = (_Float16)zs[e];
  if (e == 0) *status = 0;
}

// hd = tanh(zs @ W_d1 + b_d1): 8 waves/block, one 16x16 tile per wave, K=64 (2 WMMA)
__global__ __launch_bounds__(256) void gemm1_kernel(const _Float16* __restrict__ Afp,
                                                    const _Float16* __restrict__ BT,
                                                    const float* __restrict__ bias,
                                                    _Float16* __restrict__ out) {
  int lane = threadIdx.x & 31;
  int tile = blockIdx.x * 8 + (threadIdx.x >> 5);   // 768 tiles = 64 x 12
  int mtile = tile / 12, ntile = tile - mtile * 12;
  int ma = mtile * 16 + (lane & 15);
  int n  = ntile * 16 + (lane & 15);
  int ksub = (lane >> 4) * 8;
  v8f acc = {};
#pragma unroll
  for (int kt = 0; kt < 2; ++kt) {
    int k0 = kt * 32;
    v8h a0 = *(const v8h*)(Afp + ma * 64 + k0 + ksub * 2);
    v8h a1 = *(const v8h*)(Afp + ma * 64 + k0 + ksub * 2 + 8);
    v16h av = __builtin_shufflevector(a0, a1, 0,1,2,3,4,5,6,7,8,9,10,11,12,13,14,15);
    v16h bv = *(const v16h*)(BT + n * 64 + k0 + ksub * 2);
    acc = wmma_f16(av, bv, acc);
  }
  float b = (n < HID) ? bias[n] : 0.0f;
  int rb = mtile * 16 + ((lane >> 4) * 8);
#pragma unroll
  for (int r = 0; r < 8; ++r)
    out[(rb + r) * 192 + n] = (_Float16)tanhf(acc[r] + b);
}

// n_s = hd @ W_d2 + b_d2: 8 waves/block, K=192 (6 WMMA), write f32, n<466
__global__ __launch_bounds__(256) void gemm2_kernel(const _Float16* __restrict__ Afp,
                                                    const _Float16* __restrict__ BT,
                                                    const float* __restrict__ bias,
                                                    float* __restrict__ out) {
  int lane = threadIdx.x & 31;
  int tile = blockIdx.x * 8 + (threadIdx.x >> 5);   // 1920 tiles = 64 x 30
  int mtile = tile / 30, ntile = tile - mtile * 30;
  int ma = mtile * 16 + (lane & 15);
  int n  = ntile * 16 + (lane & 15);
  int ksub = (lane >> 4) * 8;
  v8f acc = {};
#pragma unroll
  for (int kt = 0; kt < 6; ++kt) {
    int k0 = kt * 32;
    v8h a0 = *(const v8h*)(Afp + ma * 192 + k0 + ksub * 2);
    v8h a1 = *(const v8h*)(Afp + ma * 192 + k0 + ksub * 2 + 8);
    v16h av = __builtin_shufflevector(a0, a1, 0,1,2,3,4,5,6,7,8,9,10,11,12,13,14,15);
    v16h bv = *(const v16h*)(BT + n * 192 + k0 + ksub * 2);
    acc = wmma_f16(av, bv, acc);
  }
  float b = (n < N_DIM) ? bias[n] : 0.0f;
  int rb = mtile * 16 + ((lane >> 4) * 8);
#pragma unroll
  for (int r = 0; r < 8; ++r) {
    if (n < N_DIM) out[(rb + r) * N_DIM + n] = acc[r] + b;
  }
}

// ---------------- launch ----------------

extern "C" void kernel_launch(void* const* d_in, const int* in_sizes, int n_in,
                              void* d_out, int out_size, void* d_ws, size_t ws_size,
                              hipStream_t stream) {
  const float* n0    = (const float*)d_in[0];
  const float* p     = (const float*)d_in[1];
  const float* tstep = (const float*)d_in[2];
  const float* We1   = (const float*)d_in[3];
  const float* be1   = (const float*)d_in[4];
  const float* We2   = (const float*)d_in[5];
  const float* be2   = (const float*)d_in[6];
  const float* A     = (const float*)d_in[7];
  const float* B     = (const float*)d_in[8];
  const float* Wd1   = (const float*)d_in[9];
  const float* bd1   = (const float*)d_in[10];
  const float* Wd2   = (const float*)d_in[11];
  const float* bd2   = (const float*)d_in[12];

  char* ws = (char*)d_ws;
  _Float16* Bs16 = (_Float16*)(ws + WS_BS16);
  _Float16* wd1t = (_Float16*)(ws + WS_WD1T);
  _Float16* wd2t = (_Float16*)(ws + WS_WD2T);
  _Float16* zs16 = (_Float16*)(ws + WS_ZS16);
  _Float16* hd16 = (_Float16*)(ws + WS_HD16);
  float*    z0b  = (float*)(ws + WS_Z0);

  float* out = (float*)d_out;
  float* ns  = out;
  float* zs  = out + ZS_OFF;
  int*   st  = (int*)(out + ST_OFF);

  prep_bsym_kernel<<<(64 * KPAD + 255) / 256, 256, 0, stream>>>(B, A, Bs16);
  prep_wd1_kernel<<<(192 * 64 + 255) / 256, 256, 0, stream>>>(Wd1, wd1t);
  prep_wd2_kernel<<<(480 * 192 + 255) / 256, 256, 0, stream>>>(Wd2, wd2t);
  encoder_kernel<<<1, 256, 0, stream>>>(n0, p, We1, be1, We2, be2, z0b);
  ode_kernel<<<1, 512, SMEM_BYTES, stream>>>(tstep, z0b, Bs16, zs);
  conv_zs_kernel<<<(T_PTS * Z_DIM + 255) / 256, 256, 0, stream>>>(zs, zs16, st);
  gemm1_kernel<<<96, 256, 0, stream>>>(zs16, wd1t, bd1, hd16);
  gemm2_kernel<<<240, 256, 0, stream>>>(hd16, wd2t, bd2, ns);
}
